// GraphRNNBilinear_46420006535677
// MI455X (gfx1250) — matrix operand order repaired and checked by reading, not compiled, not run on hardware
//
#include <hip/hip_runtime.h>
#include <hip/hip_bf16.h>

typedef __bf16 bf16;
typedef __attribute__((ext_vector_type(16))) __bf16 v16bf;
typedef __attribute__((ext_vector_type(8)))  __bf16 v8bf;
typedef __attribute__((ext_vector_type(8)))  float  v8f;

union FragBF { v16bf v; v8bf h[2]; };

// ---------- WMMA fragment helpers (CDNA5 16-bit layouts, CDNA5 ISA) ----------
// A (16x32, MxK): lanes 0-15 row M=lane, K 0..7 in v0..3, K 16..23 in v4..7;
// lanes 16-31 same rows, K 8..15 / 24..31.  => two b128 loads at +8*half, +16 more.
__device__ __forceinline__ v16bf load_frag_a(const bf16* A, int lda, int mbase, int kb, int lane) {
    int half = lane >> 4;
    const bf16* p = A + (size_t)(mbase + (lane & 15)) * lda + kb + 8 * half;
    FragBF f;
    f.h[0] = *(const v8bf*)(p);
    f.h[1] = *(const v8bf*)(p + 16);
    return f.v;
}
// B (32x16, KxN) built from row-major W[N,K] (B[k,n] = W[n,k]):
// lanes 0-15 hold K=kb..kb+7 / kb+16..kb+23, lanes 16-31 hold K=kb+8..kb+15 / kb+24..kb+31.
__device__ __forceinline__ v16bf load_frag_b(const bf16* W, int ldw, int nbase, int kb, int lane) {
    int half = lane >> 4;
    const bf16* p = W + (size_t)(nbase + (lane & 15)) * ldw + kb + 8 * half;
    FragBF f;
    f.h[0] = *(const v8bf*)(p);
    f.h[1] = *(const v8bf*)(p + 16);
    return f.v;
}

__device__ __forceinline__ v8f wmma_bf16(v16bf a, v16bf b, v8f c) {
    return __builtin_amdgcn_wmma_f32_16x16x32_bf16(false, a, false, b, (short)0, c, false, false);
}

// ---------- prep kernels ----------
__global__ void cvt_f32_bf16(const float* __restrict__ s, bf16* __restrict__ d, int n) {
    for (int i = blockIdx.x * blockDim.x + threadIdx.x; i < n; i += gridDim.x * blockDim.x)
        d[i] = (bf16)s[i];
}

// x_cat[b, s, :] = s==0 ? sentinel : concat(input[b,s-1,:768], tags[b,s-1,:256]); bf16 [8192,1024]
__global__ void build_x(const float* __restrict__ inp, const float* __restrict__ tag,
                        const float* __restrict__ sent, bf16* __restrict__ x) {
    const int total = 8192 * 1024;
    for (int i = blockIdx.x * blockDim.x + threadIdx.x; i < total; i += gridDim.x * blockDim.x) {
        int d = i & 1023;
        int n = i >> 10;
        int s = n & 255;
        int b = n >> 8;
        float v;
        if (s == 0)        v = sent[d];
        else if (d < 768)  v = inp[((size_t)(b * 255 + s - 1)) * 768 + d];
        else               v = tag[((size_t)(b * 255 + s - 1)) * 256 + (d - 768)];
        x[i] = (bf16)v;
    }
}

__global__ void bias_sum(const float* a0, const float* b0, float* o0,
                         const float* a1, const float* b1, float* o1) {
    int i = blockIdx.x * blockDim.x + threadIdx.x;
    if (i < 512)          o0[i] = a0[i] + b0[i];
    else if (i < 1024)    { int j = i - 512; o1[j] = a1[j] + b1[j]; }
}

__global__ void init_edge(const bf16* __restrict__ gs, bf16* __restrict__ h0,
                          bf16* __restrict__ h1, float* __restrict__ Amat) {
    const int total = 8192 * 512;
    for (int i = blockIdx.x * blockDim.x + threadIdx.x; i < total; i += gridDim.x * blockDim.x) {
        h0[i] = gs[i];
        h1[i] = (bf16)0.f;
        if (i < 8192) Amat[i * 26] = 1.0f;
    }
}

// ---------- generic WMMA GEMM ----------
// C[M,N] = act( A[M,K] * W[N,K]^T  (+ A2[M,K2] * W2[N,K2]^T)  + bias[N] + rowvec[m]*colvec[n] )
// One wave computes a 32x64 tile: 2 m-tiles x 4 n-tiles = 8 accumulators; every B fragment
// is reused by both m-tiles (halves L2 weight traffic vs 16x64).
__global__ void __launch_bounds__(256)
gemm_wmma(const bf16* __restrict__ A, int lda,
          const bf16* __restrict__ W, int ldw,
          const bf16* __restrict__ A2, int lda2,
          const bf16* __restrict__ W2, int ldw2, int K2,
          int M, int N, int K,
          const float* __restrict__ bias,
          const float* __restrict__ rowvec, int rowstride,
          const float* __restrict__ colvec,
          float* __restrict__ Cf, int ldc,
          bf16* __restrict__ Cb, int ldcb,
          int act) {
    int lane = threadIdx.x & 31;
    int wave = blockIdx.x * (blockDim.x >> 5) + (threadIdx.x >> 5);
    int tilesN = N >> 6;
    int mt = wave / tilesN;        // 32-row m-tile index
    int nt = wave % tilesN;
    if (mt * 32 >= M) return;
    int m0 = mt << 5, n0 = nt << 6;

    v8f acc[2][4];
#pragma unroll
    for (int i = 0; i < 2; ++i)
#pragma unroll
        for (int j = 0; j < 4; ++j)
#pragma unroll
            for (int r = 0; r < 8; ++r) acc[i][j][r] = 0.f;

    for (int kb = 0; kb < K; kb += 32) {
        v16bf a0 = load_frag_a(A, lda, m0, kb, lane);
        v16bf a1 = load_frag_a(A, lda, m0 + 16, kb, lane);
        __builtin_prefetch(A + (size_t)(m0 + (lane & 15)) * lda + kb + 64, 0, 0);
#pragma unroll
        for (int j = 0; j < 4; ++j) {
            v16bf b = load_frag_b(W, ldw, n0 + (j << 4), kb, lane);
            acc[0][j] = wmma_bf16(a0, b, acc[0][j]);
            acc[1][j] = wmma_bf16(a1, b, acc[1][j]);
        }
    }
    if (A2) {
        for (int kb = 0; kb < K2; kb += 32) {
            v16bf a0 = load_frag_a(A2, lda2, m0, kb, lane);
            v16bf a1 = load_frag_a(A2, lda2, m0 + 16, kb, lane);
#pragma unroll
            for (int j = 0; j < 4; ++j) {
                v16bf b = load_frag_b(W2, ldw2, n0 + (j << 4), kb, lane);
                acc[0][j] = wmma_bf16(a0, b, acc[0][j]);
                acc[1][j] = wmma_bf16(a1, b, acc[1][j]);
            }
        }
    }

    int half = lane >> 4, col = lane & 15;
#pragma unroll
    for (int j = 0; j < 4; ++j) {
        int n = n0 + (j << 4) + col;
        float bn = bias ? bias[n] : 0.f;
        float cv = colvec ? colvec[n] : 0.f;
#pragma unroll
        for (int i = 0; i < 2; ++i) {
#pragma unroll
            for (int r = 0; r < 8; ++r) {
                int m = m0 + (i << 4) + r + (half << 3);
                float v = acc[i][j][r] + bn;
                if (rowvec) v += rowvec[(size_t)m * rowstride] * cv;
                if (act == 1)      v = tanhf(v);
                else if (act == 2) v = (v > 0.f) ? v : (expf(v) - 1.f);
                if (Cf) Cf[(size_t)m * ldc + n] = v;
                if (Cb) Cb[(size_t)m * ldcb + n] = (bf16)v;
            }
        }
    }
}

// ---------- persistent tanh-RNN recurrence: h_s = tanh(xw[:,s,:] + h_{s-1} Whh^T + bhh) ----------
// One workgroup (32 waves), h double-buffered in LDS (bf16). Wave w owns n-cols [16w,16w+16),
// both m-tiles (M=32=batch). B frags stream from L2, A frags via ds_load_b128.
__global__ void __launch_bounds__(1024)
rnn_recur(const float* __restrict__ xw,   // [(b*256+s)*512 + n]
          const bf16* __restrict__ Whh,   // [512,512]
          const float* __restrict__ bhh,  // [512]
          bf16* __restrict__ yout,        // [8192,512]
          const int* __restrict__ mask) { // nullable, [32,255]
    __shared__ __align__(16) bf16 hbuf[2][32 * 512];
    int lane = threadIdx.x & 31;
    int wave = threadIdx.x >> 5;   // 0..31
    int n0 = wave << 4;
    for (int i = threadIdx.x; i < 32 * 512; i += 1024) hbuf[1][i] = (bf16)0.f;
    __syncthreads();

    int half = lane >> 4, col = lane & 15;
    for (int s = 0; s < 256; ++s) {
        int cur = s & 1, prv = cur ^ 1;
        const bf16* hp = hbuf[prv];
        v8f acc0, acc1;
#pragma unroll
        for (int r = 0; r < 8; ++r) { acc0[r] = 0.f; acc1[r] = 0.f; }
        for (int kb = 0; kb < 512; kb += 32) {
            v16bf b  = load_frag_b(Whh, 512, n0, kb, lane);
            v16bf a0 = load_frag_a(hp, 512, 0,  kb, lane);
            v16bf a1 = load_frag_a(hp, 512, 16, kb, lane);
            acc0 = wmma_bf16(a0, b, acc0);
            acc1 = wmma_bf16(a1, b, acc1);
        }
        int n = n0 + col;
        float bn = bhh[n];
#pragma unroll
        for (int r = 0; r < 8; ++r) {
            int m0 = r + (half << 3);       // 0..15
            int m1 = m0 + 16;               // 16..31
            float v0 = tanhf(xw[((size_t)(m0 * 256 + s)) * 512 + n] + acc0[r] + bn);
            float v1 = tanhf(xw[((size_t)(m1 * 256 + s)) * 512 + n] + acc1[r] + bn);
            hbuf[cur][m0 * 512 + n] = (bf16)v0;
            hbuf[cur][m1 * 512 + n] = (bf16)v1;
            float o0 = v0, o1 = v1;
            if (mask && s > 0) {
                o0 *= (float)mask[m0 * 255 + s - 1];
                o1 *= (float)mask[m1 * 255 + s - 1];
            }
            yout[((size_t)(m0 * 256 + s)) * 512 + n] = (bf16)o0;
            yout[((size_t)(m1 * 256 + s)) * 512 + n] = (bf16)o1;
        }
        __syncthreads();
    }
}

// ---------- pred[n] = dot(h1[n,:], clsW) + clsb -> Amat[n*26 + t + 1] ----------
__global__ void pred_kernel(const bf16* __restrict__ h1, const float* __restrict__ clsW,
                            const float* __restrict__ clsb, float* __restrict__ Amat, int t) {
    int row = blockIdx.x * (blockDim.x >> 5) + (threadIdx.x >> 5);
    if (row >= 8192) return;
    int lane = threadIdx.x & 31;
    const bf16* p = h1 + (size_t)row * 512;
    float s = 0.f;
    for (int i = lane; i < 512; i += 32) s += (float)p[i] * clsW[i];
#pragma unroll
    for (int off = 16; off > 0; off >>= 1) s += __shfl_xor(s, off, 32);
    if (lane == 0) Amat[row * 26 + t + 1] = s + clsb[0];
}

// ---------- arc_logits[b,i,j] = (0 <= i-j <= 25) ? Amat[(b*256+i)*26 + (i-j)] : 0 ----------
__global__ void arc_kernel(const float* __restrict__ Amat, float* __restrict__ out) {
    const int total = 32 * 256 * 256;
    for (int i = blockIdx.x * blockDim.x + threadIdx.x; i < total; i += gridDim.x * blockDim.x) {
        int b = i >> 16;
        int r = i & 65535;
        int ii = r >> 8;
        int j = r & 255;
        int k = ii - j;
        out[i] = (k >= 0 && k <= 25) ? Amat[((size_t)((b << 8) + ii)) * 26 + k] : 0.f;
    }
}

// ---------------------------------------------------------------------------
extern "C" void kernel_launch(void* const* d_in, const int* in_sizes, int n_in,
                              void* d_out, int out_size, void* d_ws, size_t ws_size,
                              hipStream_t stream) {
    (void)in_sizes; (void)n_in; (void)out_size; (void)ws_size;
    const float* input   = (const float*)d_in[0];
    const float* tags    = (const float*)d_in[1];
    const int*   mask    = (const int*)  d_in[2];
    const float* sent    = (const float*)d_in[3];
    const float* g0_Wih  = (const float*)d_in[4];
    const float* g0_Whh  = (const float*)d_in[5];
    const float* g0_bih  = (const float*)d_in[6];
    const float* g0_bhh  = (const float*)d_in[7];
    const float* g1_Wih  = (const float*)d_in[8];
    const float* g1_Whh  = (const float*)d_in[9];
    const float* g1_bih  = (const float*)d_in[10];
    const float* g1_bhh  = (const float*)d_in[11];
    const float* e0_Wih  = (const float*)d_in[12];  // [512,1] -> colvec
    const float* e0_Whh  = (const float*)d_in[13];
    const float* e0_bih  = (const float*)d_in[14];
    const float* e0_bhh  = (const float*)d_in[15];
    const float* e1_Wih  = (const float*)d_in[16];
    const float* e1_Whh  = (const float*)d_in[17];
    const float* e1_bih  = (const float*)d_in[18];
    const float* e1_bhh  = (const float*)d_in[19];
    const float* cls_W   = (const float*)d_in[20];
    const float* cls_b   = (const float*)d_in[21];
    const float* ht_W    = (const float*)d_in[22];
    const float* ht_b    = (const float*)d_in[23];
    const float* dt_W    = (const float*)d_in[24];
    const float* dt_b    = (const float*)d_in[25];

    float* out_arc  = (float*)d_out;                 // [32,256,256]
    float* out_head = out_arc + 32 * 256 * 256;      // [32,256,256]
    float* out_dep  = out_head + 32 * 256 * 256;     // [32,256,256]

    char* ws = (char*)d_ws;
    size_t off = 0;
    auto alloc = [&](size_t bytes) -> void* {
        void* p = ws + off;
        off = (off + bytes + 255) & ~(size_t)255;
        return p;
    };
    bf16* xcat  = (bf16*)alloc((size_t)8192 * 1024 * 2);
    bf16* wg0ih = (bf16*)alloc((size_t)512 * 1024 * 2);
    bf16* wg0hh = (bf16*)alloc((size_t)512 * 512 * 2);
    bf16* wg1ih = (bf16*)alloc((size_t)512 * 512 * 2);
    bf16* wg1hh = (bf16*)alloc((size_t)512 * 512 * 2);
    bf16* we0hh = (bf16*)alloc((size_t)512 * 512 * 2);
    bf16* we1ih = (bf16*)alloc((size_t)512 * 512 * 2);
    bf16* we1hh = (bf16*)alloc((size_t)512 * 512 * 2);
    bf16* wht   = (bf16*)alloc((size_t)256 * 512 * 2);
    bf16* wdt   = (bf16*)alloc((size_t)256 * 512 * 2);
    float* xw0  = (float*)alloc((size_t)8192 * 512 * 4);
    float* xw1  = (float*)alloc((size_t)8192 * 512 * 4);
    bf16* y0    = (bf16*)alloc((size_t)8192 * 512 * 2);
    bf16* gs    = (bf16*)alloc((size_t)8192 * 512 * 2);
    bf16* h0a   = (bf16*)alloc((size_t)8192 * 512 * 2);
    bf16* h0b   = (bf16*)alloc((size_t)8192 * 512 * 2);
    bf16* h1a   = (bf16*)alloc((size_t)8192 * 512 * 2);
    bf16* h1b   = (bf16*)alloc((size_t)8192 * 512 * 2);
    float* eb0  = (float*)alloc(512 * 4);
    float* eb1  = (float*)alloc(512 * 4);
    float* Amat = (float*)alloc((size_t)8192 * 26 * 4);

    auto cvt = [&](const float* s, bf16* d, int n) {
        cvt_f32_bf16<<<(n + 255) / 256, 256, 0, stream>>>(s, d, n);
    };
    cvt(g0_Wih, wg0ih, 512 * 1024);
    cvt(g0_Whh, wg0hh, 512 * 512);
    cvt(g1_Wih, wg1ih, 512 * 512);
    cvt(g1_Whh, wg1hh, 512 * 512);
    cvt(e0_Whh, we0hh, 512 * 512);
    cvt(e1_Wih, we1ih, 512 * 512);
    cvt(e1_Whh, we1hh, 512 * 512);
    cvt(ht_W,   wht,   256 * 512);
    cvt(dt_W,   wdt,   256 * 512);
    build_x<<<32768, 256, 0, stream>>>(input, tags, sent, xcat);
    bias_sum<<<4, 256, 0, stream>>>(e0_bih, e0_bhh, eb0, e1_bih, e1_bhh, eb1);

    auto gemm = [&](const bf16* A, int lda, const bf16* W, int ldw,
                    const bf16* A2, int lda2, const bf16* W2, int ldw2, int K2,
                    int M, int N, int K, const float* bias,
                    const float* rowvec, int rowstride, const float* colvec,
                    float* Cf, int ldc, bf16* Cb, int ldcb, int act) {
        int waves = (M / 32) * (N / 64);
        int blocks = (waves + 7) / 8;
        gemm_wmma<<<blocks, 256, 0, stream>>>(A, lda, W, ldw, A2, lda2, W2, ldw2, K2,
                                              M, N, K, bias, rowvec, rowstride, colvec,
                                              Cf, ldc, Cb, ldcb, act);
    };

    // layer g0: input projection then recurrence
    gemm(xcat, 1024, wg0ih, 1024, nullptr, 0, nullptr, 0, 0,
         8192, 512, 1024, g0_bih, nullptr, 0, nullptr, xw0, 512, nullptr, 0, 0);
    rnn_recur<<<1, 1024, 0, stream>>>(xw0, wg0hh, g0_bhh, y0, nullptr);
    // layer g1
    gemm(y0, 512, wg1ih, 512, nullptr, 0, nullptr, 0, 0,
         8192, 512, 512, g1_bih, nullptr, 0, nullptr, xw1, 512, nullptr, 0, 0);
    rnn_recur<<<1, 1024, 0, stream>>>(xw1, wg1hh, g1_bhh, gs, mask);

    // head/dep tags: elu(gs @ W^T + b)
    gemm(gs, 512, wht, 512, nullptr, 0, nullptr, 0, 0,
         8192, 256, 512, ht_b, nullptr, 0, nullptr, out_head, 256, nullptr, 0, 2);
    gemm(gs, 512, wdt, 512, nullptr, 0, nullptr, 0, 0,
         8192, 256, 512, dt_b, nullptr, 0, nullptr, out_dep, 256, nullptr, 0, 2);

    // edge RNN: h0 = gs, h1 = 0, xe = 1
    init_edge<<<16384, 256, 0, stream>>>(gs, h0a, h1a, Amat);
    bf16 *h0c = h0a, *h0n = h0b, *h1c = h1a, *h1n = h1b;
    for (int t = 0; t < 25; ++t) {
        // h0_new = tanh(xe*e0_Wih + (e0_bih+e0_bhh) + h0 @ e0_Whh^T)
        gemm(h0c, 512, we0hh, 512, nullptr, 0, nullptr, 0, 0,
             8192, 512, 512, eb0, Amat + t, 26, e0_Wih, nullptr, 0, h0n, 512, 1);
        // h1_new = tanh(h0_new @ e1_Wih^T + h1 @ e1_Whh^T + (e1_bih+e1_bhh))
        gemm(h0n, 512, we1ih, 512, h1c, 512, we1hh, 512, 512,
             8192, 512, 512, eb1, nullptr, 0, nullptr, nullptr, 0, h1n, 512, 1);
        pred_kernel<<<1024, 256, 0, stream>>>(h1n, cls_W, cls_b, Amat, t);
        bf16* tmp;
        tmp = h0c; h0c = h0n; h0n = tmp;
        tmp = h1c; h1c = h1n; h1n = tmp;
    }
    arc_kernel<<<8192, 256, 0, stream>>>(Amat, out_arc);
}
